// NeuromorphicCompressionLayer_19842748908226
// MI455X (gfx1250) — matrix operand order repaired and checked by reading, not compiled, not run on hardware
//
#include <hip/hip_runtime.h>

// ---------------------------------------------------------------------------
// Liquid State Machine on MI455X (gfx1250, wave32, WMMA)
//   B=16, T=256, DIN=1024, R=4096, DOUT=64
// Phase 1: in_cur[t*B+b][r] = x @ W_in^T   (bf16 WMMA, f32 acc, tile/wave)
// Phase 2: 256 stream-ordered step kernels, latency-optimized:
//   - 256 rec blocks: one 16(b)x16(r) tile each, split-K x4 across the
//     block's 4 waves (32 WMMAs/wave), LDS reduction, wave0 does LIF
//   - 4 readout blocks: previous step's out[:, t-1, :], same split-K scheme
// ---------------------------------------------------------------------------

#define DEV_INLINE __device__ __forceinline__

constexpr int kB = 16, kT = 256, kDIN = 1024, kR = 4096, kDOUT = 64;
constexpr int kMTOT = kB * kT;            // 4096 rows of input GEMM (m = t*B + b)
constexpr float kNEG_INF = -1.0e9f;

typedef __attribute__((ext_vector_type(8)))  __bf16 v8bf;
typedef __attribute__((ext_vector_type(16))) __bf16 v16bf;
typedef __attribute__((ext_vector_type(8)))  float  v8f;

// ---- workspace layout (bytes; all offsets 256B-aligned) --------------------
constexpr size_t SZ_INCUR  = sizeof(float)  * (size_t)kMTOT * kR;   // 64 MB
constexpr size_t SZ_WRES16 = sizeof(__bf16) * (size_t)kR * kR;      // 32 MB
constexpr size_t SZ_WIN16  = sizeof(__bf16) * (size_t)kR * kDIN;    //  8 MB
constexpr size_t SZ_X16    = sizeof(__bf16) * (size_t)kMTOT * kDIN; //  8 MB
constexpr size_t SZ_WOUT16 = sizeof(__bf16) * (size_t)kDOUT * kR;   // .5 MB
constexpr size_t SZ_V      = sizeof(float)  * 2ull * kB * kR;       // ping-pong
constexpr size_t SZ_V16    = sizeof(__bf16) * 2ull * kB * kR;
constexpr size_t SZ_LAST   = sizeof(float)  * (size_t)kB * kR;

constexpr size_t OFF_INCUR  = 0;
constexpr size_t OFF_WRES16 = OFF_INCUR  + SZ_INCUR;
constexpr size_t OFF_WIN16  = OFF_WRES16 + SZ_WRES16;
constexpr size_t OFF_X16    = OFF_WIN16  + SZ_WIN16;
constexpr size_t OFF_WOUT16 = OFF_X16    + SZ_X16;
constexpr size_t OFF_V      = OFF_WOUT16 + SZ_WOUT16;
constexpr size_t OFF_V16    = OFF_V      + SZ_V;
constexpr size_t OFF_LAST   = OFF_V16    + SZ_V16;

// ---------------------------------------------------------------------------
// WMMA operand loaders (bf16, 16x16x32 tile), per CDNA5 VGPR layout tables.
// A (MxK=16x32): lane L (g=L/16, m=L%16) holds K=[8g..8g+7] and [16+8g..16+8g+7]
// B (KxN=32x16): lane L (g=L/16, n=L%16) holds column n, K=[8g..8g+7] and [16+8g..16+8g+7]
// ---------------------------------------------------------------------------
DEV_INLINE v16bf load_a_bf16(const __bf16* __restrict__ A, int lda,
                             int m_base, int k0, int lane) {
  const int g = lane >> 4, mr = lane & 15;
  const __bf16* p = A + (size_t)(m_base + mr) * lda + k0 + 8 * g;
  v8bf a0 = *(const v8bf*)p;
  v8bf a1 = *(const v8bf*)(p + 16);
  v16bf a;
#pragma unroll
  for (int i = 0; i < 8; ++i) { a[i] = a0[i]; a[i + 8] = a1[i]; }
  return a;
}

DEV_INLINE v16bf load_b_bf16(const __bf16* __restrict__ Wrows, int ldb,
                             int n_base, int k0, int lane) {
  const int g = lane >> 4, n = lane & 15;
  const __bf16* p = Wrows + (size_t)(n_base + n) * ldb + k0 + 8 * g;
  v8bf b0 = *(const v8bf*)p;
  v8bf b1 = *(const v8bf*)(p + 16);
  v16bf b;
#pragma unroll
  for (int i = 0; i < 8; ++i) { b[i] = b0[i]; b[i + 8] = b1[i]; }
  return b;
}

// ---------------------------------------------------------------------------
// Elementwise prep kernels
// ---------------------------------------------------------------------------
__global__ void lsm_cvt_bf16_kernel(const float* __restrict__ src,
                                    __bf16* __restrict__ dst, int n) {
  int i = blockIdx.x * blockDim.x + threadIdx.x;
  if (i < n) dst[i] = (__bf16)src[i];
}

// x[b][t][d] (f32) -> x16[m=t*B+b][d] (bf16): GEMM-ready A layout
__global__ void lsm_permute_x_kernel(const float* __restrict__ x,
                                     __bf16* __restrict__ x16) {
  int i = blockIdx.x * blockDim.x + threadIdx.x;
  if (i >= kMTOT * kDIN) return;
  int m = i / kDIN, d = i - m * kDIN;
  int t = m >> 4, b = m & 15;                    // kB == 16
  x16[i] = (__bf16)x[((size_t)b * kT + t) * kDIN + d];
}

__global__ void lsm_init_state_kernel(float* __restrict__ v,
                                      __bf16* __restrict__ v16,
                                      float* __restrict__ last) {
  int i = blockIdx.x * blockDim.x + threadIdx.x;
  const int n = kB * kR;
  if (i >= n) return;
  v[i] = 0.f;  v[n + i] = 0.f;
  v16[i] = (__bf16)0.f;  v16[n + i] = (__bf16)0.f;
  last[i] = kNEG_INF;
}

// ---------------------------------------------------------------------------
// Phase 1: input-current GEMM.  M=kMTOT, N=kR, K=kDIN.  One 16x16 tile / wave.
// ---------------------------------------------------------------------------
__global__ __launch_bounds__(128) void lsm_input_gemm_kernel(
    const __bf16* __restrict__ x16,      // [kMTOT][kDIN]
    const __bf16* __restrict__ win16,    // [kR][kDIN]
    float* __restrict__ in_cur)          // [kMTOT][kR]
{
  const int lane = threadIdx.x & 31;
  const int wave = threadIdx.x >> 5;
  const int NT = kR / 16;                              // 256 n-tiles
  const int tile = blockIdx.x * 4 + wave;
  const int mt = tile / NT, nt = tile - mt * NT;
  const int m_base = mt * 16, n_base = nt * 16;

  const __bf16* brow = win16 + (size_t)(n_base + (lane & 15)) * kDIN;
  v8f acc = {};
  for (int k0 = 0; k0 < kDIN; k0 += 32) {
    if (k0 + 64 < kDIN) __builtin_prefetch(brow + k0 + 64, 0, 1);
    v16bf a = load_a_bf16(x16, kDIN, m_base, k0, lane);
    v16bf b = load_b_bf16(win16, kDIN, n_base, k0, lane);
    acc = __builtin_amdgcn_wmma_f32_16x16x32_bf16(false, a, false, b,
                                                  (short)0, acc, false, false);
  }
  const int g = lane >> 4, n = lane & 15;
#pragma unroll
  for (int i = 0; i < 8; ++i) {
    const int m = m_base + i + 8 * g;
    in_cur[(size_t)m * kR + n_base + n] = acc[i];
  }
}

// ---------------------------------------------------------------------------
// Split-K 16x16 tile: the block's 4 waves each cover kR/4 of K (32 WMMAs),
// partials reduced through LDS.  Returns the f32 tile sum (valid on wave 0).
// ---------------------------------------------------------------------------
DEV_INLINE v8f splitk_tile_bf16(const __bf16* __restrict__ A,      // [16][kR]
                                const __bf16* __restrict__ Brows,  // rows, ld=kR
                                int n_base, int wave, int lane,
                                float* __restrict__ red /* [4*256] */) {
  const int kq = kR / 4;
  const int klo = wave * kq, khi = klo + kq;
  const __bf16* brow = Brows + (size_t)(n_base + (lane & 15)) * kR;
  v8f acc = {};
  for (int k0 = klo; k0 < khi; k0 += 32) {
    if (k0 + 64 < khi) __builtin_prefetch(brow + k0 + 64, 0, 1);
    v16bf a = load_a_bf16(A, kR, 0, k0, lane);
    v16bf b = load_b_bf16(Brows, kR, n_base, k0, lane);
    acc = __builtin_amdgcn_wmma_f32_16x16x32_bf16(false, a, false, b,
                                                  (short)0, acc, false, false);
  }
  // stage partials: red[wave][i*32 + lane]  (conflict-free: stride-1 in lane)
#pragma unroll
  for (int i = 0; i < 8; ++i) red[wave * 256 + i * 32 + lane] = acc[i];
  __syncthreads();
  v8f sum = {};
#pragma unroll
  for (int i = 0; i < 8; ++i) {
    float s = red[i * 32 + lane];
#pragma unroll
    for (int w = 1; w < 4; ++w) s += red[w * 256 + i * 32 + lane];
    sum[i] = s;
  }
  return sum;
}

// ---------------------------------------------------------------------------
// Phase 2 step kernel (launched once per t, stream-ordered):
//   blocks 0..255   : rec tile r=[blk*16,+16) via split-K, LIF -> ping-pong
//   blocks 256..259 : readout of previous state into out[:, t-1, :]
// ---------------------------------------------------------------------------
__global__ __launch_bounds__(128) void lsm_step_kernel(
    const __bf16* __restrict__ wres16,   // [kR][kR]
    const __bf16* __restrict__ wout16,   // [kDOUT][kR]
    const float*  __restrict__ in_cur,   // [kMTOT][kR]
    const float*  __restrict__ thr,      // [kR]
    const float*  __restrict__ leak,     // [kR]
    const float*  __restrict__ refr,     // [kR]
    float*  __restrict__ vbuf,           // [2][kB][kR] ping-pong f32
    __bf16* __restrict__ v16buf,         // [2][kB][kR] ping-pong bf16
    float*  __restrict__ last,           // [kB][kR]
    float*  __restrict__ out,            // [kB][kT][kDOUT]
    int t)
{
  __shared__ float red[4 * 256];
  const int lane = threadIdx.x & 31;
  const int wave = threadIdx.x >> 5;
  const int cur = t & 1, nxt = cur ^ 1;
  const __bf16* v16 = v16buf + (size_t)cur * (kB * kR);

  const bool is_readout = (blockIdx.x >= kR / 16);     // blocks 256..259
  if (is_readout && t == 0) return;                    // no t=-1 output

  const __bf16* Brows = is_readout ? wout16 : wres16;
  const int n_base = (is_readout ? (blockIdx.x - kR / 16) : blockIdx.x) * 16;

  v8f sum = splitk_tile_bf16(v16, Brows, n_base, wave, lane, red);
  if (wave != 0) return;                               // wave-uniform branch

  const int g = lane >> 4, n = lane & 15;

  if (is_readout) {                                    // out[:, t-1, n_base+n]
#pragma unroll
    for (int i = 0; i < 8; ++i) {
      const int b = i + 8 * g;
      out[((size_t)b * kT + (t - 1)) * kDOUT + n_base + n] = sum[i];
    }
    return;
  }

  // --- LIF update (f32 state), write ping-pong f32 + bf16 -----------------
  const float* vprev  = vbuf   + (size_t)cur * (kB * kR);
  float*       vnew   = vbuf   + (size_t)nxt * (kB * kR);
  __bf16*      v16new = v16buf + (size_t)nxt * (kB * kR);

  const int r = n_base + n;
  const float th = thr[r], lk = leak[r], rf = refr[r];
  const float tf = (float)t;
#pragma unroll
  for (int i = 0; i < 8; ++i) {
    const int b = i + 8 * g;
    const size_t idx = (size_t)b * kR + r;
    const float v  = vprev[idx];
    const float lt = last[idx];
    const float I  = in_cur[((size_t)t * kB + b) * kR + r] + sum[i];
    const bool refract = (tf - lt) < rf;
    const float vu = refract ? v : (v + (-lk * v + I));
    const bool spiked = (!refract) && (vu >= th);
    const float vn = spiked ? 0.f : vu;
    vnew[idx]   = vn;
    v16new[idx] = (__bf16)vn;
    last[idx]   = spiked ? tf : lt;
  }
}

// Final readout for t = T-1 (state lives in buffer (T & 1)); 4 blocks.
__global__ __launch_bounds__(128) void lsm_final_readout_kernel(
    const __bf16* __restrict__ v16buf,
    const __bf16* __restrict__ wout16,
    float* __restrict__ out)
{
  __shared__ float red[4 * 256];
  const int lane = threadIdx.x & 31;
  const int wave = threadIdx.x >> 5;
  const __bf16* v16 = v16buf + (size_t)(kT & 1) * (kB * kR);
  const int n_base = blockIdx.x * 16;

  v8f sum = splitk_tile_bf16(v16, wout16, n_base, wave, lane, red);
  if (wave != 0) return;
  const int g = lane >> 4, n = lane & 15;
#pragma unroll
  for (int i = 0; i < 8; ++i) {
    const int b = i + 8 * g;
    out[((size_t)b * kT + (kT - 1)) * kDOUT + n_base + n] = sum[i];
  }
}

// ---------------------------------------------------------------------------
extern "C" void kernel_launch(void* const* d_in, const int* in_sizes, int n_in,
                              void* d_out, int out_size, void* d_ws, size_t ws_size,
                              hipStream_t stream) {
  (void)in_sizes; (void)n_in; (void)out_size; (void)ws_size;
  const float* x     = (const float*)d_in[0];  // [B][T][DIN]
  const float* W_in  = (const float*)d_in[1];  // [R][DIN]
  const float* W_res = (const float*)d_in[2];  // [R][R]
  const float* W_out = (const float*)d_in[3];  // [DOUT][R]
  const float* thr   = (const float*)d_in[4];  // [R]
  const float* leak  = (const float*)d_in[5];  // [R]
  const float* refr  = (const float*)d_in[6];  // [R]
  float* out = (float*)d_out;                  // [B][T][DOUT]

  char* ws = (char*)d_ws;
  float*  in_cur = (float*) (ws + OFF_INCUR);
  __bf16* wres16 = (__bf16*)(ws + OFF_WRES16);
  __bf16* win16  = (__bf16*)(ws + OFF_WIN16);
  __bf16* x16    = (__bf16*)(ws + OFF_X16);
  __bf16* wout16 = (__bf16*)(ws + OFF_WOUT16);
  float*  vbuf   = (float*) (ws + OFF_V);
  __bf16* v16buf = (__bf16*)(ws + OFF_V16);
  float*  lastt  = (float*) (ws + OFF_LAST);

  // --- prep: bf16 conversions + GEMM-friendly x permute + state init ------
  {
    int n = kR * kR;
    lsm_cvt_bf16_kernel<<<(n + 255) / 256, 256, 0, stream>>>(W_res, wres16, n);
    n = kR * kDIN;
    lsm_cvt_bf16_kernel<<<(n + 255) / 256, 256, 0, stream>>>(W_in, win16, n);
    n = kDOUT * kR;
    lsm_cvt_bf16_kernel<<<(n + 255) / 256, 256, 0, stream>>>(W_out, wout16, n);
    n = kMTOT * kDIN;
    lsm_permute_x_kernel<<<(n + 255) / 256, 256, 0, stream>>>(x, x16);
    n = kB * kR;
    lsm_init_state_kernel<<<(n + 255) / 256, 256, 0, stream>>>(vbuf, v16buf, lastt);
  }

  // --- phase 1: input currents for all timesteps (one 16x16 tile / wave) --
  {
    const int tiles = (kMTOT / 16) * (kR / 16);        // 65536
    lsm_input_gemm_kernel<<<tiles / 4, 128, 0, stream>>>(x16, win16, in_cur);
  }

  // --- phase 2: sequential scan, fused readout of previous step -----------
  const int step_blocks = kR / 16 + kDOUT / 16;        // 256 rec + 4 readout
  for (int t = 0; t < kT; ++t) {
    lsm_step_kernel<<<step_blocks, 128, 0, stream>>>(
        wres16, wout16, in_cur, thr, leak, refr, vbuf, v16buf, lastt, out, t);
  }
  lsm_final_readout_kernel<<<kDOUT / 16, 128, 0, stream>>>(v16buf, wout16, out);
}
